// VectorQuantizer_44890998178571
// MI455X (gfx1250) — compile-verified, matrix operand order checked by reading, not run on hardware
//
#include <hip/hip_runtime.h>
#include <hip/hip_bf16.h>

// ---------------------------------------------------------------------------
// VQ quantizer for MI455X (gfx1250, wave32).
//   d = ||z||^2 + ||e||^2 - 2 z.e ; argmin over K=1024 codes; gather; loss.
// Matmul (65536 x 1024 x 256) via V_WMMA_F32_16X16X32_BF16 with a two-term
// bf16 split of both operands (zh*eh + zl*eh + zh*el) for ~f32 argmin
// fidelity. B fragments are staged to LDS with async copies (ASYNCcnt),
// double-buffered, and shared by 8 waves to cut L2 traffic 8x.
// ---------------------------------------------------------------------------

typedef __attribute__((ext_vector_type(16))) __bf16 v16bf;
typedef __attribute__((ext_vector_type(8)))  float  v8f;

#define KCODES 1024
#define DDIM   256
#define HWSZ   1024          // 32*32
#define NTOT   65536         // 64 * 1024
#define NCHUNK 64            // KCODES / 16
#define NKD    8             // DDIM / 32
#define ZQ_ELEMS 16777216    // 64*256*32*32
#define LOSS_OFF 16777216
#define IDX_OFF  16777217

#define CHUNK_BYTES 16384    // (8 kd * 32 lanes * 32B) * 2 (hi+lo)
#define HALF_BYTES  8192     // hi region size within a chunk

// k index inside a 32-wide K slab for fragment element t, lane-half hv.
__device__ __forceinline__ int kmap(int t, int hv) {
    return (t < 8) ? (t + 8 * hv) : (t + 8 + 8 * hv);
}

__device__ __forceinline__ __bf16 bf_hi(float v) {
    unsigned u = __builtin_bit_cast(unsigned, v);
    return __builtin_bit_cast(__bf16, (unsigned short)(u >> 16));
}
__device__ __forceinline__ float bf_hi_f(float v) {
    unsigned u = __builtin_bit_cast(unsigned, v) & 0xffff0000u;
    return __builtin_bit_cast(float, u);
}

// Async copy 16B global -> LDS (gfx1250, ASYNCcnt-tracked, no VGPR transit).
__device__ __forceinline__ void async_cp16(unsigned lds_byte_addr,
                                           const void* gaddr) {
    asm volatile("global_load_async_to_lds_b128 %0, %1, off"
                 :: "v"(lds_byte_addr), "v"(gaddr) : "memory");
}

// ---------------------------------------------------------------------------
// Kernel 1: marshal embedding^T into exact B-fragment layout (hi/lo bf16).
// Global layout: [chunk cc]{ hi: [kd][lane][16], lo: [kd][lane][16] } so a
// chunk is one linear 16KB block (async-copy friendly) and a lane's fragment
// is one contiguous 32B read.
// ---------------------------------------------------------------------------
__global__ __launch_bounds__(256) void vq_prep_eB(const float* __restrict__ emb,
                                                  __bf16* __restrict__ eB) {
    const int cc   = blockIdx.x;          // 0..63 column chunk
    const int kd   = threadIdx.x >> 5;    // 0..7
    const int lane = threadIdx.x & 31;
    const int col  = cc * 16 + (lane & 15);
    const int hv   = lane >> 4;
    // element offsets inside chunk cc
    const int hbase = cc * (CHUNK_BYTES / 2) + (kd * 32 + lane) * 16;
    const int lbase = hbase + (HALF_BYTES / 2);
#pragma unroll
    for (int t = 0; t < 16; ++t) {
        int k = kd * 32 + kmap(t, hv);
        float v  = emb[(size_t)col * DDIM + k];
        float hf = bf_hi_f(v);
        eB[hbase + t] = bf_hi(v);
        eB[lbase + t] = bf_hi(v - hf);
    }
}

// ---------------------------------------------------------------------------
// Kernel 2: ||e_k||^2 for all codes (one wave per code).
// ---------------------------------------------------------------------------
__global__ __launch_bounds__(32) void vq_enorm(const float* __restrict__ emb,
                                               float* __restrict__ enorm) {
    const int code = blockIdx.x;
    const int lane = threadIdx.x;
    float s = 0.f;
#pragma unroll
    for (int j = 0; j < 8; ++j) {
        float v = emb[(size_t)code * DDIM + lane + 32 * j];
        s += v * v;
    }
#pragma unroll
    for (int off = 16; off > 0; off >>= 1) s += __shfl_xor(s, off);
    if (lane == 0) enorm[code] = s;
}

// ---------------------------------------------------------------------------
// Kernel 3: distances + argmin. 256 threads = 8 waves, 128 rows per WG.
// A fragments (split bf16) live in VGPRs for the whole K sweep; B chunks are
// async-copied to a double-buffered LDS staging area shared by all 8 waves.
// ---------------------------------------------------------------------------
__global__ __launch_bounds__(256) void vq_main(const float* __restrict__ z,
                                               const __bf16* __restrict__ eB,
                                               const float* __restrict__ enorm,
                                               int* __restrict__ d_idx) {
    __shared__ __align__(32) char ldsB[2 * CHUNK_BYTES];   // 32 KB double buffer

    const int tid   = threadIdx.x;
    const int lane  = tid & 31;
    const int strip = tid >> 5;                   // 0..7
    const int hv    = lane >> 4;                  // lane half
    const int mrow  = lane & 15;
    const int n0    = blockIdx.x * 128 + strip * 16;
    const int ng    = n0 + mrow;                  // global row for this lane
    const int b     = ng >> 10;
    const int hw    = ng & 1023;
    const float* zb = z + (size_t)b * DDIM * HWSZ + hw;

    const char* eBc = (const char*)eB;
    const unsigned lds0 = (unsigned)(size_t)(void*)&ldsB[0];

    // Kick off async stage of chunk 0 (4 x 16B per thread = 16KB per WG).
    {
        const char* src = eBc;   // chunk 0
#pragma unroll
        for (int j = 0; j < 4; ++j) {
            unsigned off = (unsigned)(tid + 256 * j) * 16u;
            async_cp16(lds0 + off, src + off);
        }
    }

    // Load A fragments straight from global in WMMA layout (coalesced:
    // lanes 0-15 and 16-31 each hit one contiguous 64B segment per load).
    v16bf Ah[NKD], Al[NKD];
    float sq = 0.f;
#pragma unroll
    for (int kd = 0; kd < NKD; ++kd) {
#pragma unroll
        for (int t = 0; t < 16; ++t) {
            int k = kd * 32 + kmap(t, hv);
            float v  = zb[(size_t)k * HWSZ];
            float hf = bf_hi_f(v);
            Ah[kd][t] = bf_hi(v);
            Al[kd][t] = bf_hi(v - hf);        // exact two-term split
            sq += v * v;
        }
    }
    // Row norms: lane L and L^16 jointly hold all 256 elems of row (L&15).
    float zn = sq + __shfl_xor(sq, 16);
    float znr[8];
#pragma unroll
    for (int r = 0; r < 8; ++r) znr[r] = __shfl(zn, 8 * hv + r);

    float minv[8];
    int   mini[8];
#pragma unroll
    for (int r = 0; r < 8; ++r) { minv[r] = 3.4e38f; mini[r] = 0; }

    for (int cc = 0; cc < NCHUNK; ++cc) {
        const int buf = cc & 1;
        // Prefetch-stage next chunk into the other buffer, then wait for the
        // current chunk's 4 async loads (in-order completion => <=4 left).
        if (cc + 1 < NCHUNK) {
            const char* src = eBc + (size_t)(cc + 1) * CHUNK_BYTES;
            unsigned dst = lds0 + (unsigned)(buf ^ 1) * CHUNK_BYTES;
#pragma unroll
            for (int j = 0; j < 4; ++j) {
                unsigned off = (unsigned)(tid + 256 * j) * 16u;
                async_cp16(dst + off, src + off);
            }
            asm volatile("s_wait_asynccnt 0x4" ::: "memory");
        } else {
            asm volatile("s_wait_asynccnt 0x0" ::: "memory");
        }
        __syncthreads();   // all threads' stage of chunk cc visible

        const char* lb = &ldsB[buf * CHUNK_BYTES];
        const v16bf* Bh = (const v16bf*)(lb);
        const v16bf* Bl = (const v16bf*)(lb + HALF_BYTES);

        v8f a0 = {}; v8f a1 = {}; v8f a2 = {};   // 3 independent WMMA chains
#pragma unroll
        for (int kd = 0; kd < NKD; ++kd) {
            v16bf bh = Bh[kd * 32 + lane];       // 32B ds_load per lane
            v16bf bl = Bl[kd * 32 + lane];
            a0 = __builtin_amdgcn_wmma_f32_16x16x32_bf16(false, Ah[kd], false, bh,
                                                         (short)0, a0, false, false);
            a1 = __builtin_amdgcn_wmma_f32_16x16x32_bf16(false, Al[kd], false, bh,
                                                         (short)0, a1, false, false);
            a2 = __builtin_amdgcn_wmma_f32_16x16x32_bf16(false, Ah[kd], false, bl,
                                                         (short)0, a2, false, false);
        }
        const int col = cc * 16 + mrow;
        const float en = enorm[col];
#pragma unroll
        for (int r = 0; r < 8; ++r) {
            float dv = znr[r] + en - 2.0f * (a0[r] + a1[r] + a2[r]);
            if (dv < minv[r]) { minv[r] = dv; mini[r] = col; }
        }
        __syncthreads();   // buffer consumed; next iter may overwrite it
    }

    // Argmin across the 16 lanes of each half (first-min tie-break).
#pragma unroll
    for (int off = 1; off < 16; off <<= 1) {
#pragma unroll
        for (int r = 0; r < 8; ++r) {
            float ov = __shfl_xor(minv[r], off);
            int   oi = __shfl_xor(mini[r], off);
            if (ov < minv[r] || (ov == minv[r] && oi < mini[r])) {
                minv[r] = ov; mini[r] = oi;
            }
        }
    }
    if (mrow == 0) {
#pragma unroll
        for (int r = 0; r < 8; ++r) d_idx[n0 + 8 * hv + r] = mini[r];
    }
}

// ---------------------------------------------------------------------------
// Kernel 4: gather z_q (straight-through output == gathered codes), emit
// indices as float, and per-block partial sums of (z_q - z)^2.
// ---------------------------------------------------------------------------
__global__ __launch_bounds__(256) void vq_gather(const float* __restrict__ z,
                                                 const float* __restrict__ emb,
                                                 const int* __restrict__ d_idx,
                                                 float* __restrict__ out,
                                                 float* __restrict__ partials) {
    __shared__ float red[256];
    const int tid = blockIdx.x * 256 + threadIdx.x;   // flat over [B,C,H,W]
    const int hw  = tid & 1023;
    const int c   = (tid >> 10) & 255;
    const int b   = tid >> 18;
    const int n   = (b << 10) + hw;
    const int idx = d_idx[n];
    const float val = emb[(size_t)idx * DDIM + c];    // L2-resident gather
    const float zv  = z[tid];                         // coalesced
    out[tid] = val;                                   // z_q, [B,C,H,W]
    if (c == 0) out[IDX_OFF + n] = (float)idx;        // indices (as f32)
    const float diff = val - zv;
    red[threadIdx.x] = diff * diff;
    __syncthreads();
#pragma unroll
    for (int s = 128; s > 0; s >>= 1) {
        if (threadIdx.x < s) red[threadIdx.x] += red[threadIdx.x + s];
        __syncthreads();
    }
    if (threadIdx.x == 0) partials[blockIdx.x] = red[0];
}

// ---------------------------------------------------------------------------
// Kernel 5: deterministic final loss reduction.
// loss = vq + 0.25*commit = 1.25 * mean((z_q - z)^2)  (stop_grad is a no-op
// on values).
// ---------------------------------------------------------------------------
__global__ __launch_bounds__(256) void vq_reduce(const float* __restrict__ partials,
                                                 float* __restrict__ out) {
    __shared__ float red[256];
    float s = 0.f;
    for (int i = threadIdx.x; i < 65536; i += 256) s += partials[i];
    red[threadIdx.x] = s;
    __syncthreads();
#pragma unroll
    for (int k = 128; k > 0; k >>= 1) {
        if (threadIdx.x < k) red[threadIdx.x] += red[threadIdx.x + k];
        __syncthreads();
    }
    if (threadIdx.x == 0)
        out[LOSS_OFF] = 1.25f * red[0] / (float)ZQ_ELEMS;
}

// ---------------------------------------------------------------------------
extern "C" void kernel_launch(void* const* d_in, const int* in_sizes, int n_in,
                              void* d_out, int out_size, void* d_ws, size_t ws_size,
                              hipStream_t stream) {
    (void)in_sizes; (void)n_in; (void)out_size; (void)ws_size;
    const float* z   = (const float*)d_in[0];   // [64,256,32,32] f32
    const float* emb = (const float*)d_in[1];   // [1024,256] f32
    float* out = (float*)d_out;

    char* ws = (char*)d_ws;
    __bf16* eB     = (__bf16*)(ws);                                  // 1 MB
    float*  enorm  = (float*)(ws + (1024u << 10));                   //   4 KB
    int*    idx    = (int*)(ws + (1024u << 10) + 4096u);             // 256 KB
    float*  parts  = (float*)(ws + (1024u << 10) + 4096u + (256u << 10)); // 256 KB

    vq_prep_eB<<<NCHUNK, 256, 0, stream>>>(emb, eB);
    vq_enorm  <<<KCODES, 32, 0, stream>>>(emb, enorm);
    vq_main   <<<NTOT / 128, 256, 0, stream>>>(z, eB, enorm, idx);
    vq_gather <<<ZQ_ELEMS / 256, 256, 0, stream>>>(z, emb, idx, out, parts);
    vq_reduce <<<1, 256, 0, stream>>>(parts, out);
}
